// Attention_70995809403431
// MI455X (gfx1250) — matrix operand order repaired and checked by reading, not compiled, not run on hardware
//
#include <hip/hip_runtime.h>

typedef __attribute__((ext_vector_type(16))) _Float16 v16h;
typedef __attribute__((ext_vector_type(8)))  _Float16 v8h;
typedef __attribute__((ext_vector_type(8)))  float    v8f;

#define SEQ    2048
#define BATCH  4
#define NHEAD  16
#define DHEAD  64
#define DMODEL 1024

static __device__ __forceinline__ v16h mk16(v8h a, v8h b) {
  v16h r;
#pragma unroll
  for (int i = 0; i < 8; ++i) { r[i] = a[i]; r[i + 8] = b[i]; }
  return r;
}

// Issue one 16-byte async global->LDS DMA (ASYNCcnt-tracked) per lane.
static __device__ __forceinline__ void async_copy_b128(const void* gsrc,
                                                       void* lds_dst) {
  unsigned lds = (unsigned)(unsigned long long)(uintptr_t)lds_dst; // low 32b = LDS offset
  unsigned long long ga = (unsigned long long)(uintptr_t)gsrc;
  asm volatile("global_load_async_to_lds_b128 %0, %1, off"
               :: "v"(lds), "v"(ga) : "memory");
}

// ---------------- f32 -> f16 conversion ----------------
__global__ void cvt_f32_f16(const float* __restrict__ src,
                            _Float16* __restrict__ dst, int n) {
  int i = blockIdx.x * blockDim.x + threadIdx.x;
  int stride = gridDim.x * blockDim.x;
  for (; i < n; i += stride) dst[i] = (_Float16)src[i];
}

// ---------------- generic WMMA GEMM ----------------
// A: [M=8192, 1024] f16 row-major.  Bw: [1024, ldb] f16 row-major.
// Block tile 128x64, 8 waves of 32x32, K-step 32 via LDS.
// MODE 0: write Q   -> [b,h,n,d] f16
// MODE 1: write K/V -> K:[b,h,n,d], V(col>=1024):[b,h,d,n] (transposed) f16
// MODE 2: write f32 out + bias
template <int MODE>
__global__ __launch_bounds__(256)
void gemm_wmma(const _Float16* __restrict__ A, const _Float16* __restrict__ Bw,
               int ldb,
               _Float16* __restrict__ outQ, _Float16* __restrict__ outK,
               _Float16* __restrict__ outVt, float* __restrict__ outF,
               const float* __restrict__ bias)
{
  __shared__ _Float16 ldsA[128 * 40];   // [row][k], padded stride 40
  __shared__ _Float16 ldsB[64 * 40];    // transposed [n][k], padded stride 40

  const int tid  = threadIdx.x;
  const int lane = tid & 31;
  const int wave = tid >> 5;
  const int wm   = wave & 3;            // 4 waves along M
  const int wn   = wave >> 2;           // 2 waves along N
  const bool lo  = (lane < 16);
  const int n16  = lane & 15;
  const int hi8  = lo ? 0 : 8;
  const int mBase = blockIdx.x * 128;
  const int nBase = blockIdx.y * 64;

  const int aRow = tid >> 2;            // staging coords
  const int aCol = (tid & 3) * 8;
  const int bK   = tid >> 3;
  const int bN8  = (tid & 7) * 8;

  v8f c[2][2] = {};

  for (int k0 = 0; k0 < DMODEL; k0 += 32) {
    __syncthreads();   // previous iteration's LDS reads complete
    // prefetch next K-step tiles into L2 while this step streams
    if (k0 + 32 < DMODEL) {
      __builtin_prefetch(A + (size_t)(mBase + aRow) * DMODEL + k0 + 32 + aCol, 0, 3);
      __builtin_prefetch(Bw + (size_t)(k0 + 32 + bK) * ldb + nBase + bN8, 0, 3);
    }
    // stage A: 128x32 halfs via async global->LDS DMA (16B per lane)
#pragma unroll
    for (int r = 0; r < 2; ++r) {
      const _Float16* src =
          A + (size_t)(mBase + aRow + r * 64) * DMODEL + k0 + aCol;
      async_copy_b128(src, &ldsA[(aRow + r * 64) * 40 + aCol]);
    }
    // stage B transposed: read [k][n] 16B, scatter to [n][k]
    {
      v8h v = *(const v8h*)(Bw + (size_t)(k0 + bK) * ldb + nBase + bN8);
#pragma unroll
      for (int i = 0; i < 8; ++i) ldsB[(bN8 + i) * 40 + bK] = v[i];
    }
    asm volatile("s_wait_asynccnt 0x0" ::: "memory");
    __syncthreads();   // all waves' DMA + ds stores visible

    v16h a[2], b[2];
#pragma unroll
    for (int mi = 0; mi < 2; ++mi) {
      int row = wm * 32 + mi * 16 + n16;
      int kb  = lo ? 0 : 8;                       // A 16x32 lane layout
      v8h a0 = *(const v8h*)&ldsA[row * 40 + kb];
      v8h a1 = *(const v8h*)&ldsA[row * 40 + kb + 16];
      a[mi] = mk16(a0, a1);
    }
#pragma unroll
    for (int ni = 0; ni < 2; ++ni) {
      int n  = wn * 32 + ni * 16 + n16;
      int kb = lo ? 0 : 8;                        // B 32x16 lane layout
      v8h b0 = *(const v8h*)&ldsB[n * 40 + kb];
      v8h b1 = *(const v8h*)&ldsB[n * 40 + kb + 16];
      b[ni] = mk16(b0, b1);
    }
#pragma unroll
    for (int mi = 0; mi < 2; ++mi)
#pragma unroll
      for (int ni = 0; ni < 2; ++ni)
        c[mi][ni] = __builtin_amdgcn_wmma_f32_16x16x32_f16(
            false, a[mi], false, b[ni], (short)0, c[mi][ni], false, false);
  }

  // ---- epilogue: C layout lane (M = v + hi8, N = n16) ----
  // A 128-row block tile never crosses a batch boundary -> b is uniform.
  const int bb = mBase >> 11;               // / SEQ
  const int srowBase = (mBase & (SEQ - 1)) + wm * 32;

#pragma unroll
  for (int mi = 0; mi < 2; ++mi)
#pragma unroll
    for (int ni = 0; ni < 2; ++ni) {
      const int srow0 = srowBase + mi * 16 + hi8;       // + v per element
      const int gn    = nBase + wn * 32 + ni * 16 + n16;
      if constexpr (MODE == 2) {
        const int gm0 = mBase + wm * 32 + mi * 16 + hi8;
        const float bv = bias[gn];
#pragma unroll
        for (int v = 0; v < 8; ++v)
          outF[(size_t)(gm0 + v) * DMODEL + gn] = c[mi][ni][v] + bv;
      } else {
        const bool isV = (MODE == 1) && ((nBase + wn * 32 + ni * 16) >= DMODEL);
        if (!isV) {
          const int h = gn >> 6, d = gn & 63;           // K or Q layout
          _Float16* dst = (MODE == 0 ? outQ : outK) +
              (((size_t)bb * NHEAD + h) * SEQ + srow0) * DHEAD + d;
#pragma unroll
          for (int v = 0; v < 8; ++v)
            dst[(size_t)v * DHEAD] = (_Float16)c[mi][ni][v];
        } else {
          const int g2 = gn - DMODEL;
          const int h = g2 >> 6, d = g2 & 63;           // V transposed layout
          _Float16* dst = outVt +
              (((size_t)bb * NHEAD + h) * DHEAD + d) * SEQ + srow0;
#pragma unroll
          for (int v = 0; v < 8; ++v)
            dst[v] = (_Float16)c[mi][ni][v];
        }
      }
    }
}

// ---------------- fused causal flash attention ----------------
// One wave per 16 Q rows; online softmax in registers; P goes through a
// per-wave LDS tile only for the C->A layout shuffle.
__global__ __launch_bounds__(256)
void flash_attn(const _Float16* __restrict__ Q, const _Float16* __restrict__ K,
                const _Float16* __restrict__ Vt, _Float16* __restrict__ ctx)
{
  __shared__ _Float16 pTile[8 * 16 * 40];   // per-wave 16x32 P tile, pad 40
  const int tid  = threadIdx.x;
  const int lane = tid & 31;
  const int wave = tid >> 5;
  _Float16* pT = &pTile[wave * 16 * 40];
  const bool lo = (lane < 16);
  const int n16 = lane & 15;
  const int hi8 = lo ? 0 : 8;
  const float INF = __builtin_inff();

  int wid  = blockIdx.x * 8 + wave;
  int qblk = wid & 127;                  // SEQ/16 blocks per (b,h)
  int h    = (wid >> 7) & (NHEAD - 1);
  int b    = wid >> 11;
  int qbase = qblk * 16;

  const _Float16* qptr = Q  + (((size_t)b * NHEAD + h) * SEQ + qbase) * DHEAD;
  const _Float16* kptr = K  + ((size_t)b * NHEAD + h) * SEQ * DHEAD;
  const _Float16* vptr = Vt + ((size_t)b * NHEAD + h) * DHEAD * SEQ;

  // Q as two A fragments (d = 0..31, 32..63)
  v16h aq[2];
#pragma unroll
  for (int t = 0; t < 2; ++t) {
    const _Float16* p = qptr + n16 * DHEAD + t * 32 + (lo ? 0 : 8);
    aq[t] = mk16(*(const v8h*)p, *(const v8h*)(p + 16));
  }

  v8f o[4] = {};
  float m_run[8], l_run[8];
#pragma unroll
  for (int v = 0; v < 8; ++v) { m_run[v] = -INF; l_run[v] = 0.0f; }

  const float scale = 0.125f;            // 1/sqrt(64)
  const int jend = qbase + 16;
  for (int jb = 0; jb < jend; jb += 32) {
    // prefetch next K chunk while computing this one
    __builtin_prefetch(kptr + (size_t)(jb + 32 + n16) * DHEAD, 0, 3);

    // S = Q @ K^T over 32 columns (two 16-col C frags)
    v8f s[2];
#pragma unroll
    for (int f = 0; f < 2; ++f) {
      v8f acc = {};
#pragma unroll
      for (int t = 0; t < 2; ++t) {
        const _Float16* p =
            kptr + (size_t)(jb + f * 16 + n16) * DHEAD + t * 32 + (lo ? 0 : 8);
        v16h bk = mk16(*(const v8h*)p, *(const v8h*)(p + 16));
        acc = __builtin_amdgcn_wmma_f32_16x16x32_f16(
            false, aq[t], false, bk, (short)0, acc, false, false);
      }
      s[f] = acc;
    }

    // scale, causal mask, online softmax stats (per-row = per 16-lane group)
    float nm[8];
#pragma unroll
    for (int v = 0; v < 8; ++v) {
      int rowg = qbase + v + hi8;
      int c0 = jb + n16, c1 = jb + 16 + n16;
      float s0 = s[0][v] * scale; if (c0 > rowg) s0 = -INF;
      float s1 = s[1][v] * scale; if (c1 > rowg) s1 = -INF;
      s[0][v] = s0; s[1][v] = s1;
      float t = fmaxf(s0, s1);
#pragma unroll
      for (int off = 8; off >= 1; off >>= 1)
        t = fmaxf(t, __shfl_xor(t, off, 32));
      nm[v] = fmaxf(m_run[v], t);
    }
#pragma unroll
    for (int v = 0; v < 8; ++v) {
      float alpha = __expf(m_run[v] - nm[v]);
      m_run[v] = nm[v];
      float p0 = __expf(s[0][v] - nm[v]);
      float p1 = __expf(s[1][v] - nm[v]);
      float r = p0 + p1;
#pragma unroll
      for (int off = 8; off >= 1; off >>= 1) r += __shfl_xor(r, off, 32);
      l_run[v] = l_run[v] * alpha + r;
      pT[(v + hi8) * 40 + n16]      = (_Float16)p0;
      pT[(v + hi8) * 40 + 16 + n16] = (_Float16)p1;
#pragma unroll
      for (int cc = 0; cc < 4; ++cc) o[cc][v] *= alpha;
    }
    asm volatile("s_wait_dscnt 0x0" ::: "memory");

    // P back as A fragment (16x32)
    v16h ap;
    {
      int kb = lo ? 0 : 8;
      const _Float16* p = &pT[n16 * 40 + kb];
      ap = mk16(*(const v8h*)p, *(const v8h*)(p + 16));
    }
    // O += P @ V  (Vt rows are contiguous over n -> clean B fragments)
#pragma unroll
    for (int cc = 0; cc < 4; ++cc) {
      const _Float16* p =
          vptr + (size_t)(cc * 16 + n16) * SEQ + jb + (lo ? 0 : 8);
      v16h bv = mk16(*(const v8h*)p, *(const v8h*)(p + 16));
      o[cc] = __builtin_amdgcn_wmma_f32_16x16x32_f16(
          false, ap, false, bv, (short)0, o[cc], false, false);
    }
  }

  // normalize and write context row-major [B*SEQ, DMODEL] f16
#pragma unroll
  for (int v = 0; v < 8; ++v) {
    float inv = 1.0f / l_run[v];
    size_t rowAddr =
        ((size_t)b * SEQ + qbase + v + hi8) * DMODEL + (size_t)h * DHEAD;
#pragma unroll
    for (int cc = 0; cc < 4; ++cc)
      ctx[rowAddr + cc * 16 + n16] = (_Float16)(o[cc][v] * inv);
  }
}

// ---------------- launcher ----------------
extern "C" void kernel_launch(void* const* d_in, const int* in_sizes, int n_in,
                              void* d_out, int out_size, void* d_ws,
                              size_t ws_size, hipStream_t stream) {
  const float* x   = (const float*)d_in[0];
  const float* Wq  = (const float*)d_in[1];
  const float* Wkv = (const float*)d_in[2];
  const float* Wo  = (const float*)d_in[3];
  const float* bo  = (const float*)d_in[4];
  float* out = (float*)d_out;

  char* ws = (char*)d_ws;
  size_t off = 0;
  auto alloc = [&](size_t bytes) -> char* {
    char* p = ws + off;
    off += (bytes + 255) & ~(size_t)255;
    return p;
  };
  _Float16* xh   = (_Float16*)alloc((size_t)BATCH * SEQ * DMODEL * 2);
  _Float16* Wqh  = (_Float16*)alloc((size_t)DMODEL * DMODEL * 2);
  _Float16* Wkvh = (_Float16*)alloc((size_t)DMODEL * 2 * DMODEL * 2);
  _Float16* Woh  = (_Float16*)alloc((size_t)DMODEL * DMODEL * 2);
  _Float16* qh   = (_Float16*)alloc((size_t)BATCH * NHEAD * SEQ * DHEAD * 2);
  _Float16* kh   = (_Float16*)alloc((size_t)BATCH * NHEAD * SEQ * DHEAD * 2);
  _Float16* vth  = (_Float16*)alloc((size_t)BATCH * NHEAD * DHEAD * SEQ * 2);
  _Float16* ctxh = (_Float16*)alloc((size_t)BATCH * SEQ * DMODEL * 2);

  cvt_f32_f16<<<4096, 256, 0, stream>>>(x, xh, BATCH * SEQ * DMODEL);
  cvt_f32_f16<<<1024, 256, 0, stream>>>(Wq, Wqh, DMODEL * DMODEL);
  cvt_f32_f16<<<2048, 256, 0, stream>>>(Wkv, Wkvh, DMODEL * 2 * DMODEL);
  cvt_f32_f16<<<1024, 256, 0, stream>>>(Wo, Woh, DMODEL * DMODEL);

  dim3 blk(256);
  // Q projection: 8192 x 1024
  gemm_wmma<0><<<dim3(64, 16), blk, 0, stream>>>(xh, Wqh, DMODEL, qh, nullptr,
                                                 nullptr, nullptr, nullptr);
  // KV projection: 8192 x 2048 (K normal, V transposed)
  gemm_wmma<1><<<dim3(64, 32), blk, 0, stream>>>(xh, Wkvh, 2 * DMODEL, nullptr,
                                                 kh, vth, nullptr, nullptr);
  // fused causal attention: 4*16*128 row-blocks, 8 waves/block
  flash_attn<<<dim3(1024), blk, 0, stream>>>(qh, kh, vth, ctxh);
  // output projection + bias: 8192 x 1024 f32
  gemm_wmma<2><<<dim3(64, 16), blk, 0, stream>>>(ctxh, Woh, DMODEL, nullptr,
                                                 nullptr, nullptr, out, bo);
}
